// PocketGNN_20761871909532
// MI455X (gfx1250) — matrix operand order repaired and checked by reading, not compiled, not run on hardware
//
#include <hip/hip_runtime.h>
#include <hip/hip_bf16.h>

// PocketGNN forward for MI455X (gfx1250, wave32).
// Heavy GEMMs: v_wmma_f32_16x16x32_bf16 (bf16 A/B, fp32 accum). A tiles are
// staged into double-buffered LDS with GLOBAL_LOAD_ASYNC_TO_LDS_B128
// (ASYNCcnt-tracked, zero VGPR staging), overlapping the WMMA chain of the
// current K-tile. B operand per-wave with RB reuse hints. Gather/row pointers
// hoisted out of the K loop (strength-reduced address streams).
//
// Dims: B=64, N=512, NB=20, PRE=1280, H=128, DEPTH=3.  BN = B*N = 32768.

#define BN 32768
#define HD 128
#define NNB 20

#if defined(__has_builtin)
#if __has_builtin(__builtin_amdgcn_global_load_async_to_lds_b128) && \
    __has_builtin(__builtin_amdgcn_s_wait_asynccnt)
#define HAVE_ASYNC_LDS 1
#endif
#endif
#ifndef HAVE_ASYNC_LDS
#define HAVE_ASYNC_LDS 0
#endif

typedef __attribute__((ext_vector_type(16))) __bf16 v16bf;
typedef __attribute__((ext_vector_type(8)))  float  v8f;
typedef int v4i __attribute__((vector_size(16)));   // matches builtin param type

union FragU {
  v16bf v;
  unsigned short u[16];
  uint4 q[2];
};

// fp32 -> bf16 via native fptrunc (RNE); backend can use hw cvt ops.
__device__ __forceinline__ unsigned short f2bf(float f) {
  __bf16 b = (__bf16)f;
  return __builtin_bit_cast(unsigned short, b);
}

// 16-byte global -> LDS copy. Async (ASYNCcnt) when available; else sync.
__device__ __forceinline__ void copy16_g2l(const unsigned short* gsrc, unsigned short* ldst) {
#if HAVE_ASYNC_LDS
  __builtin_amdgcn_global_load_async_to_lds_b128(
      (__attribute__((address_space(1))) v4i*)gsrc,
      (__attribute__((address_space(3))) v4i*)ldst, 0, 0);
#else
  *(uint4*)ldst = *(const uint4*)gsrc;
#endif
}
__device__ __forceinline__ void async_wait0() {
#if HAVE_ASYNC_LDS
  __builtin_amdgcn_s_wait_asynccnt(0);
#endif
}

// A fragment: 16x32 (MxK) bf16 tile. lane = row (lane&15); per-lane K indices
// are two contiguous runs [8h,8h+8) and [16+8h,16+8h+8), h = lane>>4.
__device__ __forceinline__ void load_fragA_bf16(FragU& f, const unsigned short* row, int half) {
  f.q[0] = *(const uint4*)(row + half * 8);
  f.q[1] = *(const uint4*)(row + 16 + half * 8);
}
// B fragment: 32x16 (KxN) bf16 tile. lane = column (lane&15); per-lane K indices
// are two runs [8h,8h+8) and [16+8h,16+8h+8) from W^T[N][K].
__device__ __forceinline__ void load_fragB(FragU& f, const unsigned short* col, int half) {
  f.q[0] = *(const uint4*)(col + half * 8);
  f.q[1] = *(const uint4*)(col + 16 + half * 8);
}

// WMMA chain over a preloaded fragment group; reuse_b hint (compile-time) on
// all but the last WMMA of the whole tile set.
template<int I, int G, int BASE, int TOT>
struct MMChain {
  static __device__ __forceinline__ void run(const FragU* a, const FragU& b, v8f* acc) {
    acc[BASE + I] = __builtin_amdgcn_wmma_f32_16x16x32_bf16(
        false, a[I].v, false, b.v, (short)0, acc[BASE + I],
        false, (BASE + I + 1 < TOT));
    MMChain<I + 1, G, BASE, TOT>::run(a, b, acc);
  }
};
template<int G, int BASE, int TOT>
struct MMChain<G, G, BASE, TOT> {
  static __device__ __forceinline__ void run(const FragU*, const FragU&, v8f*) {}
};

// Load G A-fragments (batched ds_loads), then run G WMMAs.
// LDS row stride = 40 halfwords (bank-spread padding).
template<int BASE, int G, int TOT>
__device__ __forceinline__ void tileGroup(const unsigned short* ldsRow, int half,
                                          const FragU& b, v8f* acc) {
  FragU a[G];
#pragma unroll
  for (int g = 0; g < G; ++g)
    load_fragA_bf16(a[g], ldsRow + (BASE + g) * 16 * 40, half);
  MMChain<0, G, BASE, TOT>::run(a, b, acc);
}

// ---------------------------------------------------------------------------
// Tiled WMMA GEMM:  out[M][128] = act( [A0 | A1][M][K0+K1] @ BT^T + bias )
//   A0: [M][K0] (fp32 if A0F32 else bf16), A1: [M][K1] bf16 (optional concat)
//   BT: pre-transposed bf16 weights [128][K0+K1]
//   ACT: 0 = leaky-relu(0.1), 1 = relu
//   RES: outF[m][n] += act(...), refresh outB (bf16)
// Block = 256 threads (8 waves) = 128 rows x 128 cols; wave w owns cols
// [16w,16w+16). A tile double-buffered in LDS: bf16 sources use async
// global->LDS copies overlapped with the WMMAs; fp32 (pre0) stages through
// registers with conversion. One barrier per K-tile.
// ---------------------------------------------------------------------------
template<int K0, int K1, int A0F32, int ACT, int RES, int HF, int HF2, int HB>
__global__ __launch_bounds__(256, 1) void gemm128_kernel(
    const void* __restrict__ A0, const unsigned short* __restrict__ A1,
    const unsigned short* __restrict__ BT,
    const float* __restrict__ bias,
    float* __restrict__ outF, float* __restrict__ outF2,
    unsigned short* __restrict__ outB)
{
  constexpr int KTOT = K0 + K1;
  constexpr int NKT = KTOT / 32;
  __shared__ unsigned short ldsA[2][128 * 40];   // ping-pong, 128 rows x 32 hw

  const int tid  = threadIdx.x;
  const int lane = tid & 31, wave = tid >> 5;
  const int half = lane >> 4, l16 = lane & 15;
  const int mBlk = blockIdx.x * 128;

  // Per-chunk loop-invariant pointers (rr = c>>2, p = c&3; 8 halfwords each).
  const float*          fPtr[2];   // A0F32 path
  const unsigned short* aPtr[2];   // A0 bf16 path (kb = 0)
  const unsigned short* a1Ptr[2];  // A1 path (kb = K0)
  int dOff[2];
#pragma unroll
  for (int i = 0; i < 2; ++i) {
    const int c = tid + i * 256;
    const int rr = c >> 2, p = c & 3;
    dOff[i] = rr * 40 + p * 8;
    if (A0F32) fPtr[i] = (const float*)A0 + (size_t)(mBlk + rr) * K0 + p * 8;
    else       aPtr[i] = (const unsigned short*)A0 + (size_t)(mBlk + rr) * K0 + p * 8;
    if (K1 > 0) a1Ptr[i] = A1 + (size_t)(mBlk + rr) * K1 + p * 8;
  }

  // ---- fp32 register staging (pre0 path only) --------------------------
  float4 sf[2][2];
  auto loadChunks = [&](int kt) {
#pragma unroll
    for (int i = 0; i < 2; ++i) {
      const float* s = fPtr[i] + kt * 32;
      sf[i][0] = *(const float4*)s;
      sf[i][1] = *(const float4*)(s + 4);
    }
  };
  auto storeChunks = [&](unsigned short* buf) {
#pragma unroll
    for (int i = 0; i < 2; ++i) {
      unsigned short* d = buf + dOff[i];
      d[0] = f2bf(sf[i][0].x); d[1] = f2bf(sf[i][0].y);
      d[2] = f2bf(sf[i][0].z); d[3] = f2bf(sf[i][0].w);
      d[4] = f2bf(sf[i][1].x); d[5] = f2bf(sf[i][1].y);
      d[6] = f2bf(sf[i][1].z); d[7] = f2bf(sf[i][1].w);
    }
  };
  // ---- bf16 async staging ----------------------------------------------
  auto issueStage = [&](int kt, unsigned short* buf) {
    const int kb = kt * 32;
#pragma unroll
    for (int i = 0; i < 2; ++i) {
      const unsigned short* src =
          (K1 == 0 || kb < K0) ? aPtr[i] + kb : a1Ptr[i] + (kb - K0);
      copy16_g2l(src, buf + dOff[i]);
    }
  };

  const v8f vzero = {0.f, 0.f, 0.f, 0.f, 0.f, 0.f, 0.f, 0.f};
  v8f acc[8];
#pragma unroll
  for (int i = 0; i < 8; ++i) acc[i] = vzero;

  const unsigned short* btRow = BT + (size_t)(wave * 16 + l16) * KTOT;

  if (A0F32) { loadChunks(0); storeChunks(&ldsA[0][0]); }
  else       { issueStage(0, &ldsA[0][0]); async_wait0(); }
  __syncthreads();

  for (int kt = 0; kt < NKT; ++kt) {
    FragU b;
    load_fragB(b, btRow + kt * 32, half);       // this wave's 16 cols only
    unsigned short* nxt = &ldsA[(kt + 1) & 1][0];
    if (kt + 1 < NKT) {
      if (A0F32) loadChunks(kt + 1);            // global -> regs
      else       issueStage(kt + 1, nxt);       // async global -> LDS (other buf)
    }
    const unsigned short* ldsRow = &ldsA[kt & 1][0] + l16 * 40;
    tileGroup<0, 4, 8>(ldsRow, half, b, acc);   // 4 ds-load pairs, 4 WMMAs
    tileGroup<4, 4, 8>(ldsRow, half, b, acc);
    if (A0F32 && kt + 1 < NKT) storeChunks(nxt);
    if (!A0F32) async_wait0();
    __syncthreads();
  }

  // Epilogue. C layout: VGPR v, lane L -> m = mBlk + 16t + v + 8*(L>>4),
  // n = wave*16 + (L&15)
  const int n = wave * 16 + l16;
  const float bb = bias[n];
#pragma unroll
  for (int t = 0; t < 8; ++t) {
#pragma unroll
    for (int v = 0; v < 8; ++v) {
      const int m = mBlk + t * 16 + v + half * 8;
      float x = acc[t][v] + bb;
      x = (x > 0.f) ? x : (ACT == 0 ? 0.1f * x : 0.f);
      const size_t o = (size_t)m * HD + n;
      if (RES) {
        const float hnew = outF[o] + x;
        outF[o] = hnew;
        if (HB) outB[o] = f2bf(hnew);
      } else {
        if (HF)  outF[o]  = x;
        if (HF2) outF2[o] = x;
        if (HB)  outB[o]  = f2bf(x);
      }
    }
  }
}

// ---------------------------------------------------------------------------
// Fused gather + Wga GEMM + masked neighbor reduction.
// Block = 8 nodes = 160 gathered rows; A row m=(node,nb):
//   k in [0,32)   -> edgebf[bond_nb[m]]   (enei)
//   k in [32,160) -> hbf[atom_nb[m]]      (nei)
// Gather row pointers precomputed once per thread; per K-tile the async
// global->LDS copies (double-buffered) overlap the 10-WMMA chain.
// ver=relu(acc+bias), t[node] = relu(sum_nb ver*mask) -> tbf (bf16).
// LDS: [0,12800) bufA0 | [12800,25600) bufA1 | [0,40960) ver scratch
// (aliases the A buffers; only used after the K loop) | mask at 40960.
// ---------------------------------------------------------------------------
#define WGA_BUF_HW (160 * 40)
#define WGA_SHMEM  (40960 + 160 * 4)

__global__ __launch_bounds__(256, 1) void wga_gather_gemm_kernel(
    const unsigned short* __restrict__ hbf,     // [BN][128] bf16
    const unsigned short* __restrict__ edgebf,  // [BN][32] bf16
    const int* __restrict__ atom_nb,            // [BN*NNB]
    const int* __restrict__ bond_nb,            // [BN*NNB]
    const float* __restrict__ nbs_mask,         // [BN*NNB]
    const unsigned short* __restrict__ WgaT,    // [128][160] bf16
    const float* __restrict__ bias,             // [128]
    unsigned short* __restrict__ tbf)           // [BN][128] bf16
{
  extern __shared__ char smem[];
  unsigned short* ldsA0 = (unsigned short*)smem;
  unsigned short* ldsA1 = (unsigned short*)(smem + WGA_BUF_HW * 2);
  float* ldsV = (float*)smem;                   // aliases A buffers (post-loop)
  float* ldsM = (float*)(smem + 40960);

  const int tid  = threadIdx.x;
  const int lane = tid & 31, wave = tid >> 5;
  const int half = lane >> 4, l16 = lane & 15;
  const int ndBase = blockIdx.x * 8;          // 8 nodes per block
  const int bbat = ndBase >> 9;               // batch id (block never straddles)

  for (int r = tid; r < 160; r += 256) {
    ldsM[r] = nbs_mask[ndBase * NNB + r];     // visible by first loop barrier
  }

  // Per-thread gather pointers (up to 3 chunks of 160 rows x 4 chunks).
  // hPtr[i] points at k-tile kt=1 position; stage kt adds (kt-1)*32.
  const unsigned short* hPtr[3];
  int dOff[3];
#pragma unroll
  for (int i = 0; i < 3; ++i) {
    const int c = tid + i * 256;
    if (c < 640) {
      const int rr = c >> 2, p = c & 3;
      const int g = ndBase * NNB + rr;
      hPtr[i] = hbf + ((size_t)((bbat << 9) | atom_nb[g])) * HD + p * 8;
      dOff[i] = rr * 40 + p * 8;
    }
  }

  // Prologue: stage kt=0 (edge gather; pointers used once, not kept live).
#pragma unroll
  for (int i = 0; i < 3; ++i) {
    const int c = tid + i * 256;
    if (c < 640) {
      const int rr = c >> 2, p = c & 3;
      const unsigned short* src =
          edgebf + ((size_t)((bbat << 9) | bond_nb[ndBase * NNB + rr])) * 32 + p * 8;
      copy16_g2l(src, ldsA0 + dOff[i]);
    }
  }

  const v8f vzero = {0.f, 0.f, 0.f, 0.f, 0.f, 0.f, 0.f, 0.f};
  v8f acc[10];
#pragma unroll
  for (int t = 0; t < 10; ++t) acc[t] = vzero;

  const unsigned short* btRow = WgaT + (size_t)(wave * 16 + l16) * 160;

  async_wait0();
  __syncthreads();

  for (int kt = 0; kt < 5; ++kt) {
    FragU b;
    load_fragB(b, btRow + kt * 32, half);        // this wave's 16 cols
    unsigned short* nxt = (kt & 1) ? ldsA0 : ldsA1;
    if (kt + 1 < 5) {                            // async gather into other buf
#pragma unroll
      for (int i = 0; i < 3; ++i) {
        const int c = tid + i * 256;
        if (c < 640) copy16_g2l(hPtr[i] + (size_t)kt * 32, nxt + dOff[i]);
      }
    }
    const unsigned short* ldsRow = ((kt & 1) ? ldsA1 : ldsA0) + l16 * 40;
    tileGroup<0, 5, 10>(ldsRow, half, b, acc);   // 5 ds-load pairs, 5 WMMAs
    tileGroup<5, 5, 10>(ldsRow, half, b, acc);
    async_wait0();
    __syncthreads();
  }

  // Masked-relu dump + neighbor reduction (two half-tile passes of 4 nodes).
  const float bcol = bias[wave * 16 + l16];
  float* vv = ldsV + wave * (80 * 16);
#pragma unroll
  for (int p = 0; p < 2; ++p) {
#pragma unroll
    for (int t = 0; t < 5; ++t) {
      const int tt = p * 5 + t;
#pragma unroll
      for (int v = 0; v < 8; ++v) {
        const int r = tt * 16 + v + half * 8;       // block-row in [p*80, p*80+80)
        float x = acc[tt][v] + bcol;
        x = x > 0.f ? x : 0.f;                      // relu(ver)
        vv[(r - p * 80) * 16 + l16] = x * ldsM[r];
      }
    }
    __syncthreads();
#pragma unroll
    for (int jj = 0; jj < 2; ++jj) {
      const int jn = half * 2 + jj;                 // local node 0..3
      float s = 0.f;
#pragma unroll
      for (int r = 0; r < NNB; ++r) s += vv[(jn * NNB + r) * 16 + l16];
      s = s > 0.f ? s : 0.f;                        // relu(t)
      tbf[(size_t)(ndBase + p * 4 + jn) * HD + wave * 16 + l16] = f2bf(s);
    }
    __syncthreads();
  }
}

// ---------------------------------------------------------------------------
// Small layers (scalar; negligible FLOPs):
//   surf = sur1(lrelu(sur0(p_sur)))  -> bf16 [BN][32]
//   edge = edge_w(p_edge)            -> bf16 [BN][32]
// ---------------------------------------------------------------------------
__global__ __launch_bounds__(256) void embed_small_kernel(
    const float* __restrict__ psur, const float* __restrict__ pedge,
    const float* __restrict__ s0w, const float* __restrict__ s0b,
    const float* __restrict__ s1w, const float* __restrict__ s1b,
    const float* __restrict__ ew,  const float* __restrict__ eb,
    unsigned short* __restrict__ surfbf, unsigned short* __restrict__ edgebf)
{
  const int m = blockIdx.x * blockDim.x + threadIdx.x;
  if (m >= BN) return;
  float s0[32];
  const float* xs = psur + (size_t)m * 16;
#pragma unroll 4
  for (int o = 0; o < 32; ++o) {
    float a = s0b[o];
    for (int k = 0; k < 16; ++k) a += xs[k] * s0w[k * 32 + o];
    s0[o] = a > 0.f ? a : 0.1f * a;
  }
#pragma unroll 4
  for (int o = 0; o < 32; ++o) {
    float a = s1b[o];
    for (int k = 0; k < 32; ++k) a += s0[k] * s1w[k * 32 + o];
    surfbf[(size_t)m * 32 + o] = f2bf(a);
  }
  const float* xe = pedge + (size_t)m * 8;
#pragma unroll 4
  for (int o = 0; o < 32; ++o) {
    float a = eb[o];
    for (int k = 0; k < 8; ++k) a += xe[k] * ew[k * 32 + o];
    edgebf[(size_t)m * 32 + o] = f2bf(a);
  }
}

// Transpose-pack fp32 weight [K][N] -> bf16 W^T [N][K]
__global__ void pack_wT_kernel(const float* __restrict__ w, int K, int N,
                               unsigned short* __restrict__ out) {
  const int i = blockIdx.x * blockDim.x + threadIdx.x;
  if (i >= K * N) return;
  const int n = i / K, k = i % K;
  out[i] = f2bf(w[(size_t)k * N + n]);
}

// hpbf = bf16(h + h0)
__global__ void addh0_kernel(const float* __restrict__ a, const float* __restrict__ b,
                             unsigned short* __restrict__ o, int n) {
  const int i = blockIdx.x * blockDim.x + threadIdx.x;
  if (i < n) o[i] = f2bf(a[i] + b[i]);
}

// ---------------------------------------------------------------------------
// Host launcher. Param order = jax pytree flatten (sorted keys, 'b' before 'w'):
// Wg[0..2](b,w), Wga[0..2](b,w), edge, fin0, fin1, pre0, pre1, sur0, sur1.
// ---------------------------------------------------------------------------
extern "C" void kernel_launch(void* const* d_in, const int* in_sizes, int n_in,
                              void* d_out, int out_size, void* d_ws, size_t ws_size,
                              hipStream_t stream) {
  (void)in_sizes; (void)n_in; (void)out_size; (void)ws_size;
  const float* p_pre    = (const float*)d_in[0];
  const float* p_sur    = (const float*)d_in[1];
  const float* p_edge   = (const float*)d_in[2];
  const int*   atom_nb  = (const int*)d_in[3];
  const int*   bond_nb  = (const int*)d_in[4];
  const float* nbs_mask = (const float*)d_in[5];
  const float* Wg_b[3]  = {(const float*)d_in[6],  (const float*)d_in[8],  (const float*)d_in[10]};
  const float* Wg_w[3]  = {(const float*)d_in[7],  (const float*)d_in[9],  (const float*)d_in[11]};
  const float* Wga_b[3] = {(const float*)d_in[12], (const float*)d_in[14], (const float*)d_in[16]};
  const float* Wga_w[3] = {(const float*)d_in[13], (const float*)d_in[15], (const float*)d_in[17]};
  const float* edge_b = (const float*)d_in[18];
  const float* edge_w = (const float*)d_in[19];
  const float* fin0_b = (const float*)d_in[20];
  const float* fin0_w = (const float*)d_in[21];
  const float* fin1_b = (const float*)d_in[22];
  const float* fin1_w = (const float*)d_in[23];
  const float* pre0_b = (const float*)d_in[24];
  const float* pre0_w = (const float*)d_in[25];
  const float* pre1_b = (const float*)d_in[26];
  const float* pre1_w = (const float*)d_in[27];
  const float* sur0_b = (const float*)d_in[28];
  const float* sur0_w = (const float*)d_in[29];
  const float* sur1_b = (const float*)d_in[30];
  const float* sur1_w = (const float*)d_in[31];

  char* ws = (char*)d_ws;
  size_t off = 0;
  auto alloc = [&](size_t bytes) {
    char* p = ws + off;
    off = (off + bytes + 255) & ~(size_t)255;
    return p;
  };
  float* h            = (float*)alloc((size_t)BN * HD * 4);
  float* h0           = (float*)alloc((size_t)BN * HD * 4);
  unsigned short* hbf    = (unsigned short*)alloc((size_t)BN * HD * 2);
  unsigned short* tmpbf  = (unsigned short*)alloc((size_t)BN * HD * 2);
  unsigned short* tbf    = (unsigned short*)alloc((size_t)BN * HD * 2);
  unsigned short* hpbf   = (unsigned short*)alloc((size_t)BN * HD * 2);
  unsigned short* surfbf = (unsigned short*)alloc((size_t)BN * 32 * 2);
  unsigned short* edgebf = (unsigned short*)alloc((size_t)BN * 32 * 2);
  unsigned short* pre0T  = (unsigned short*)alloc(1280 * 128 * 2);
  unsigned short* pre1T  = (unsigned short*)alloc(128 * 128 * 2);
  unsigned short* fin0T  = (unsigned short*)alloc(160 * 128 * 2);
  unsigned short* fin1T  = (unsigned short*)alloc(128 * 128 * 2);
  unsigned short* WgT[3];
  unsigned short* WgaT[3];
  for (int d = 0; d < 3; ++d) WgT[d]  = (unsigned short*)alloc(256 * 128 * 2);
  for (int d = 0; d < 3; ++d) WgaT[d] = (unsigned short*)alloc(160 * 128 * 2);

  auto pack = [&](const float* w, int K, int N, unsigned short* dst) {
    const int tot = K * N;
    pack_wT_kernel<<<(tot + 255) / 256, 256, 0, stream>>>(w, K, N, dst);
  };
  pack(pre0_w, 1280, 128, pre0T);
  pack(pre1_w, 128, 128, pre1T);
  pack(fin0_w, 160, 128, fin0T);
  pack(fin1_w, 128, 128, fin1T);
  for (int d = 0; d < 3; ++d) { pack(Wg_w[d], 256, 128, WgT[d]); pack(Wga_w[d], 160, 128, WgaT[d]); }

  embed_small_kernel<<<BN / 256, 256, 0, stream>>>(
      p_sur, p_edge, sur0_w, sur0_b, sur1_w, sur1_b, edge_w, edge_b, surfbf, edgebf);

  // h = lrelu(pre1(lrelu(pre0 @ p_pre)))  -> h (f32), h0 (f32 copy), hbf (bf16)
  gemm128_kernel<1280, 0, 1, 0, 0, 0, 0, 1><<<BN / 128, 256, 0, stream>>>(
      p_pre, nullptr, pre0T, pre0_b, nullptr, nullptr, tmpbf);
  gemm128_kernel<128, 0, 0, 0, 0, 1, 1, 1><<<BN / 128, 256, 0, stream>>>(
      tmpbf, nullptr, pre1T, pre1_b, h, h0, hbf);

  for (int d = 0; d < 3; ++d) {
    wga_gather_gemm_kernel<<<BN / 8, 256, WGA_SHMEM, stream>>>(
        hbf, edgebf, atom_nb, bond_nb, nbs_mask, WgaT[d], Wga_b[d], tbf);
    // h += relu(Wg @ [h|t]);  hbf refreshed
    gemm128_kernel<128, 128, 0, 1, 1, 0, 0, 1><<<BN / 128, 256, 0, stream>>>(
        hbf, tbf, WgT[d], Wg_b[d], h, nullptr, hbf);
  }

  addh0_kernel<<<(BN * HD) / 256, 256, 0, stream>>>(h, h0, hpbf, BN * HD);
  // out = lrelu(fin1(lrelu(fin0 @ [h+h0 | surf])))
  gemm128_kernel<128, 32, 0, 0, 0, 0, 0, 1><<<BN / 128, 256, 0, stream>>>(
      hpbf, surfbf, fin0T, fin0_b, nullptr, nullptr, tmpbf);
  gemm128_kernel<128, 0, 0, 0, 0, 1, 0, 0><<<BN / 128, 256, 0, stream>>>(
      tmpbf, nullptr, fin1T, fin1_b, (float*)d_out, nullptr, nullptr);
}
